// AttentionHead_38895223833301
// MI455X (gfx1250) — matrix operand order repaired and checked by reading, not compiled, not run on hardware
//
#include <hip/hip_runtime.h>
#include <hip/hip_bf16.h>

// ---------------------------------------------------------------------------
// AttentionHead on MI455X (gfx1250): bf16 WMMA flash attention.
//   B=8, T=2048, D=1024, H=128.
//   - pre-pass converts W (and x if ws allows) to bf16 once
//   - QKV projection: v_wmma_f32_16x16x32_bf16, 16x128 tile per wave
//   - flash attention: DOUBLE-BUFFERED async global->LDS staging (ASYNCcnt,
//     s_wait_asynccnt 8 overlaps next-tile fetch with current-tile WMMAs)
// ---------------------------------------------------------------------------

#define B_ 8
#define T_ 2048
#define D_ 1024
#define H_ 128

typedef __attribute__((ext_vector_type(16))) __bf16 v16bf;
typedef __attribute__((ext_vector_type(8)))  float  v8f;
typedef __attribute__((ext_vector_type(4)))  float  f4;
typedef __attribute__((ext_vector_type(4)))  unsigned int u4;

union V16 { v16bf bf; u4 u[2]; };
union V8H { __bf16 h[8]; u4 u; };

__device__ __forceinline__ v8f wmma_bf16(v16bf a, v16bf b, v8f c) {
  return __builtin_amdgcn_wmma_f32_16x16x32_bf16(
      /*neg_a=*/false, a, /*neg_b=*/false, b,
      /*c_mod=*/(short)0, c, /*reuse_a=*/false, /*reuse_b=*/false);
}

// Async global->LDS copy of 16 bytes per lane (ASYNCcnt-tracked, gfx1250).
__device__ __forceinline__ void async_b128_to_lds(const void* gsrc, void* lds_dst) {
  unsigned lds_addr = (unsigned)(size_t)lds_dst;   // low 32 bits = LDS address
  asm volatile("global_load_async_to_lds_b128 %0, %1, off"
               :: "v"(lds_addr), "v"(gsrc) : "memory");
}

// ---------------------------------------------------------------------------
// Elementwise f32 -> bf16 converter (n must be a multiple of 8).
// ---------------------------------------------------------------------------
__global__ __launch_bounds__(256) void cvt_f32_bf16_kernel(
    const float* __restrict__ src, __bf16* __restrict__ dst, int n)
{
  const int i = (blockIdx.x * 256 + threadIdx.x) * 8;
  if (i >= n) return;
  f4 a = *(const f4*)(src + i);
  f4 b = *(const f4*)(src + i + 4);
  V8H o;
  #pragma unroll
  for (int e = 0; e < 4; ++e) { o.h[e] = (__bf16)a[e]; o.h[4 + e] = (__bf16)b[e]; }
  *(u4*)(dst + i) = o.u;
}

// ---------------------------------------------------------------------------
// Kernel 1: QKV projection.  grid = (B*T/64, 3), block = 128 (4 waves).
//   mode 0: q -> qT[b][h][t]   (transposed, feeds WMMA B-matrix in attention)
//   mode 1: k -> kbf[b][t][h]  (row-major, feeds WMMA A-matrix)
//   mode 2: v -> vbf[b][t][h]  (row-major, feeds WMMA B-matrix)
// W is pre-converted bf16; x is bf16 if XBF else f32 (converted in-register).
// ---------------------------------------------------------------------------
template <bool XBF>
__global__ __launch_bounds__(128) void qkv_proj_kernel(
    const float* __restrict__ xf, const __bf16* __restrict__ xb,
    const __bf16* __restrict__ Wbf,    // 3 x [D][H] bf16, mode-major
    const float* __restrict__ bq, const float* __restrict__ bk,
    const float* __restrict__ bv,
    __bf16* __restrict__ qT, __bf16* __restrict__ kbf, __bf16* __restrict__ vbf)
{
  const int mode = blockIdx.y;
  const __bf16* W   = Wbf + (size_t)mode * D_ * H_;
  const float* bias = (mode == 0) ? bq : (mode == 1) ? bk : bv;

  const int lane  = threadIdx.x & 31;
  const int wave  = threadIdx.x >> 5;
  const int row0  = blockIdx.x * 64 + wave * 16;      // flat row in [0, B*T)
  const int r     = row0 + (lane & 15);               // A-matrix row for lane
  const int kbase = (lane < 16) ? 0 : 8;              // A-layout split-K base
  const int mofs  = (lane < 16) ? 0 : 8;              // C-layout row offset

  v8f acc[8];
  #pragma unroll
  for (int nt = 0; nt < 8; ++nt) acc[nt] = v8f{};

  for (int ks = 0; ks < D_ / 32; ++ks) {
    // ---- A tile (documented 16-bit A layout) ----
    v16bf a;
    if constexpr (XBF) {
      const __bf16* ap = xb + (size_t)r * D_ + ks * 32 + kbase;
      V16 t;
      t.u[0] = *(const u4*)(ap);        // K = kbase+0..7
      t.u[1] = *(const u4*)(ap + 16);   // K = kbase+16..23
      a = t.bf;
    } else {
      const float* ap = xf + (size_t)r * D_ + ks * 32 + kbase;
      f4 a0 = *(const f4*)(ap);
      f4 a1 = *(const f4*)(ap + 4);
      f4 a2 = *(const f4*)(ap + 16);
      f4 a3 = *(const f4*)(ap + 20);
      #pragma unroll
      for (int e = 0; e < 4; ++e) {
        a[e]      = (__bf16)a0[e];  a[4 + e]  = (__bf16)a1[e];
        a[8 + e]  = (__bf16)a2[e];  a[12 + e] = (__bf16)a3[e];
      }
    }
    // ---- B tiles: bf16 W (lane = N column, elements = K in table order) ----
    const __bf16* bp = W + (size_t)(ks * 32 + kbase) * H_ + (lane & 15);
    #pragma unroll
    for (int nt = 0; nt < 8; ++nt) {
      V16 bm;
      #pragma unroll
      for (int e = 0; e < 16; ++e) bm.bf[e] = bp[(size_t)(e + (e & 8)) * H_ + nt * 16];
      acc[nt] = wmma_bf16(a, bm.bf, acc[nt]);
    }
  }

  // ---- bias + store (C layout: VGPR i, lanes0-15 -> M=i, lanes16-31 -> M=i+8)
  #pragma unroll
  for (int nt = 0; nt < 8; ++nt) {
    const int h  = nt * 16 + (lane & 15);
    const float bb = bias[h];
    #pragma unroll
    for (int i = 0; i < 8; ++i) {
      const float val  = acc[nt][i] + bb;
      const int   rowm = row0 + mofs + i;             // flat b*T + t
      if (mode == 0) {
        const int bi = rowm >> 11;                    // / T_
        const int ti = rowm & (T_ - 1);
        qT[((size_t)bi * H_ + h) * T_ + ti] = (__bf16)val;
      } else if (mode == 1) {
        kbf[(size_t)rowm * H_ + h] = (__bf16)val;
      } else {
        vbf[(size_t)rowm * H_ + h] = (__bf16)val;
      }
    }
  }
}

// ---------------------------------------------------------------------------
// Kernel 2: causal flash attention.  grid = (T/64, B), block = 128 (4 waves).
// scores[t,s] = k_t . q_s  =>  flash "query" = k rows, "key" = q, values = v.
// q/v tiles double-buffered in LDS; async loads for step it+1 overlap the
// WMMA + softmax work of step it (s_wait_asynccnt 8 keeps them in flight).
// ---------------------------------------------------------------------------
__global__ __launch_bounds__(128) void flash_attn_kernel(
    const __bf16* __restrict__ qT,   // [b][h][t]
    const __bf16* __restrict__ kbf,  // [b][t][h]
    const __bf16* __restrict__ vbf,  // [b][t][h]
    float* __restrict__ out)         // [b][t][h] f32
{
  __shared__ __bf16 lds_q[2][128 * 32];  // qT tile: [d=0..127][s=0..31]  2x8 KB
  __shared__ __bf16 lds_v[2][32 * 128];  // v tile:  [s=0..31][h=0..127]  2x8 KB
  __shared__ __bf16 lds_p[4 * 16 * 32];  // per-wave P scratch               4 KB

  const int b     = blockIdx.y;
  const int t0b   = blockIdx.x * 64;
  const int lane  = threadIdx.x & 31;
  const int wave  = threadIdx.x >> 5;
  const int t0w   = t0b + wave * 16;
  const int mofs  = (lane < 16) ? 0 : 8;
  const int kbase = (lane < 16) ? 0 : 8;

  // issue the 8 async b128 loads staging tile [s0, s0+32) into buffer buf
  auto stage = [&](int s0, int buf) {
    const __bf16* srcq = qT + ((size_t)(b * H_ + threadIdx.x)) * T_ + s0;
    __bf16* dstq = &lds_q[buf][threadIdx.x * 32];
    #pragma unroll
    for (int i = 0; i < 4; ++i) async_b128_to_lds(srcq + 8 * i, dstq + 8 * i);
    const int rr = threadIdx.x >> 2, qq = threadIdx.x & 3;
    const __bf16* srcv = vbf + ((size_t)(b * T_ + s0 + rr)) * H_ + qq * 32;
    __bf16* dstv = &lds_v[buf][rr * 128 + qq * 32];
    #pragma unroll
    for (int i = 0; i < 4; ++i) async_b128_to_lds(srcv + 8 * i, dstv + 8 * i);
  };

  // ---- K rows resident in A-layout registers for the whole s-loop ----
  v16bf kreg[4];
  {
    const __bf16* kp = kbf + ((size_t)(b * T_ + t0w + (lane & 15))) * H_;
    #pragma unroll
    for (int ks = 0; ks < 4; ++ks) {
      V16 t;
      t.u[0] = *(const u4*)(kp + ks * 32 + kbase);
      t.u[1] = *(const u4*)(kp + ks * 32 + kbase + 16);
      kreg[ks] = t.bf;
    }
  }

  v8f oacc[8];
  #pragma unroll
  for (int nt = 0; nt < 8; ++nt) oacc[nt] = v8f{};
  float mrun[8], lrun[8];
  #pragma unroll
  for (int i = 0; i < 8; ++i) { mrun[i] = -1e30f; lrun[i] = 0.0f; }

  const float scale = 0.08838834764831845f;   // H^-0.5
  const int   nIter = t0b / 32 + 2;           // covers s <= t0b + 63

  stage(0, 0);                                // prologue: prefetch tile 0

  for (int it = 0; it < nIter; ++it) {
    const int cur = it & 1;
    const int s0  = it * 32;

    if (it + 1 < nIter) {
      // prefetch next tile into the other buffer (its readers finished at the
      // end-of-compute barrier of iteration it-1), then wait only for the
      // CURRENT tile's 8 loads: async loads complete in order per wave.
      stage(s0 + 32, cur ^ 1);
      asm volatile("s_wait_asynccnt 8" ::: "memory");
    } else {
      asm volatile("s_wait_asynccnt 0" ::: "memory");
    }
    __syncthreads();   // everyone's current tile is resident

    // ---- S = K x Q^T : two 16x16 N-tiles over 4 K-steps of d ----
    v8f st0 = v8f{}, st1 = v8f{};
    #pragma unroll
    for (int ks = 0; ks < 4; ++ks) {
      const __bf16* q0 = &lds_q[cur][(ks * 32 + kbase) * 32 + (lane & 15)];
      V16 b0, b1;
      #pragma unroll
      for (int e = 0; e < 16; ++e) {
        b0.bf[e] = q0[(e + (e & 8)) * 32];
        b1.bf[e] = q0[(e + (e & 8)) * 32 + 16];
      }
      st0 = wmma_bf16(kreg[ks], b0.bf, st0);
      st1 = wmma_bf16(kreg[ks], b1.bf, st1);
    }

    // ---- causal mask + online softmax (row = VGPR slot, col = lane&15) ----
    float alpha[8];
    #pragma unroll
    for (int i = 0; i < 8; ++i) {
      const int trow = t0w + mofs + i;
      const int n0   = s0 + (lane & 15);
      float a0 = (n0      <= trow) ? st0[i] * scale : -1e30f;
      float a1 = (n0 + 16 <= trow) ? st1[i] * scale : -1e30f;
      float mx = fmaxf(a0, a1);
      #pragma unroll
      for (int off = 1; off < 16; off <<= 1)
        mx = fmaxf(mx, __shfl_xor(mx, off, 32));      // stays in 16-lane half
      const float mnew = fmaxf(mrun[i], mx);
      const float p0 = __expf(a0 - mnew);
      const float p1 = __expf(a1 - mnew);
      st0[i] = p0; st1[i] = p1;
      float rs = p0 + p1;
      #pragma unroll
      for (int off = 1; off < 16; off <<= 1)
        rs += __shfl_xor(rs, off, 32);
      alpha[i] = __expf(mrun[i] - mnew);
      lrun[i]  = lrun[i] * alpha[i] + rs;
      mrun[i]  = mnew;
    }
    #pragma unroll
    for (int nt = 0; nt < 8; ++nt)
      #pragma unroll
      for (int i = 0; i < 8; ++i) oacc[nt][i] *= alpha[i];

    // ---- P: C-layout -> LDS -> A-layout (per-wave region) ----
    __bf16* pl = lds_p + wave * (16 * 32);
    #pragma unroll
    for (int i = 0; i < 8; ++i) {
      const int m = mofs + i;
      pl[m * 32 + (lane & 15)]      = (__bf16)st0[i];
      pl[m * 32 + 16 + (lane & 15)] = (__bf16)st1[i];
    }
    asm volatile("s_wait_dscnt 0" ::: "memory");   // LDS visible across lanes
    __builtin_amdgcn_wave_barrier();

    V16 pa;
    const __bf16* pp = pl + (lane & 15) * 32 + kbase;
    pa.u[0] = *(const u4*)(pp);
    pa.u[1] = *(const u4*)(pp + 16);

    // ---- O += P x V : 8 N-tiles across h, single K-step of 32 s ----
    #pragma unroll
    for (int nt = 0; nt < 8; ++nt) {
      const __bf16* vp = &lds_v[cur][kbase * 128 + nt * 16 + (lane & 15)];
      V16 bm;
      #pragma unroll
      for (int e = 0; e < 16; ++e) bm.bf[e] = vp[(e + (e & 8)) * 128];
      oacc[nt] = wmma_bf16(pa.bf, bm.bf, oacc[nt]);
    }

    __syncthreads();   // all reads of buffer `cur` done -> re-stageable
  }

  // ---- epilogue: out = acc / l ----
  #pragma unroll
  for (int nt = 0; nt < 8; ++nt) {
    const int h = nt * 16 + (lane & 15);
    #pragma unroll
    for (int i = 0; i < 8; ++i) {
      const int t = t0w + mofs + i;
      out[((size_t)(b * T_ + t)) * H_ + h] = oacc[nt][i] / lrun[i];
    }
  }
}

// ---------------------------------------------------------------------------
extern "C" void kernel_launch(void* const* d_in, const int* in_sizes, int n_in,
                              void* d_out, int out_size, void* d_ws, size_t ws_size,
                              hipStream_t stream) {
  const float* x  = (const float*)d_in[0];
  const float* Wq = (const float*)d_in[1];
  const float* bq = (const float*)d_in[2];
  const float* Wk = (const float*)d_in[3];
  const float* bk = (const float*)d_in[4];
  const float* Wv = (const float*)d_in[5];
  const float* bv = (const float*)d_in[6];
  float* out = (float*)d_out;

  const size_t nBTH = (size_t)B_ * T_ * H_;   // 2,097,152
  const size_t nWH  = (size_t)D_ * H_;        //   131,072 per weight
  const size_t nX   = (size_t)B_ * T_ * D_;   // 16,777,216

  __bf16* qT  = (__bf16*)d_ws;                // 4 MB
  __bf16* kbf = qT  + nBTH;                   // 4 MB
  __bf16* vbf = kbf + nBTH;                   // 4 MB
  __bf16* Wbf = vbf + nBTH;                   // 768 KB (3 weights)
  __bf16* xbf = Wbf + 3 * nWH;                // 32 MB (optional)

  const size_t base_bytes = (3 * nBTH + 3 * nWH) * sizeof(__bf16);
  const bool   use_xbf    = ws_size >= base_bytes + nX * sizeof(__bf16);

  // ---- pre-convert weights (tiny) and optionally x ----
  cvt_f32_bf16_kernel<<<(int)(nWH / 8 / 256), 256, 0, stream>>>(Wq, Wbf + 0 * nWH, (int)nWH);
  cvt_f32_bf16_kernel<<<(int)(nWH / 8 / 256), 256, 0, stream>>>(Wk, Wbf + 1 * nWH, (int)nWH);
  cvt_f32_bf16_kernel<<<(int)(nWH / 8 / 256), 256, 0, stream>>>(Wv, Wbf + 2 * nWH, (int)nWH);
  if (use_xbf)
    cvt_f32_bf16_kernel<<<(int)(nX / 8 / 256), 256, 0, stream>>>(x, xbf, (int)nX);

  dim3 pgrid(B_ * T_ / 64, 3);
  if (use_xbf)
    qkv_proj_kernel<true><<<pgrid, 128, 0, stream>>>(x, xbf, Wbf, bq, bk, bv,
                                                     qT, kbf, vbf);
  else
    qkv_proj_kernel<false><<<pgrid, 128, 0, stream>>>(x, xbf, Wbf, bq, bk, bv,
                                                      qT, kbf, vbf);

  dim3 fgrid(T_ / 64, B_);
  flash_attn_kernel<<<fgrid, 128, 0, stream>>>(qT, kbf, vbf, out);
}
